// SpikingNeuralSubstrate_36344013258758
// MI455X (gfx1250) — matrix operand order repaired and checked by reading, not compiled, not run on hardware
//
#include <hip/hip_runtime.h>
#include <stdint.h>

// Problem constants (match reference)
#define B_DIM 8192
#define N_DIM 1024
#define K_DIM 1024

typedef __attribute__((ext_vector_type(16))) __bf16 v16bf;
typedef __attribute__((ext_vector_type(8)))  __bf16 v8bf;
typedef __attribute__((ext_vector_type(8)))  float  v8f;
typedef __attribute__((ext_vector_type(4)))  float  v4f;

// Global (addrspace 1) pointer types: force global_load/global_store instead
// of flat_* (FLAT burns DScnt + aperture checks; pointer-array selection
// otherwise defeats clang's infer-address-space pass).
#define AS1 __attribute__((address_space(1)))
typedef const AS1 float* gcf;
typedef       AS1 float* gf;
typedef const AS1 v4f*   gcv4;
typedef       AS1 v4f*   gv4;

#define BM 128   // C-tile rows (batch)
#define BN 128   // C-tile cols (neurons)
#define BK 32    // K step (one bf16 WMMA K)
#define LDT 40   // padded LDS row stride in bf16 elems (80B = 5*16B: 16B aligned, bank-spread)

union Frag16 { v16bf f; v8bf h[2]; };
union Pack8  { __bf16 e[8]; v8bf v; };

// ---------------------------------------------------------------------------
// s_s = LIF(v_s, ext_sensory): pure elementwise, float4 vectorized.
// ---------------------------------------------------------------------------
__global__ __launch_bounds__(256)
void snn_lif_sensory(const float* __restrict__ ext,
                     const float* __restrict__ v,
                     float* __restrict__ out)
{
    size_t i = ((size_t)blockIdx.x * blockDim.x + threadIdx.x) * 4;
    gcf extg = (gcf)(uintptr_t)ext;
    gcf vg   = (gcf)(uintptr_t)v;
    gf  og   = (gf)(uintptr_t)out;
    v4f e  = *(gcv4)(extg + i);
    v4f vv = *(gcv4)(vg + i);
    v4f vn = vv + 0.05f * (e - vv);
    v4f s;
    s.x = (vn.x >= 1.0f) ? 1.0f : 0.0f;
    s.y = (vn.y >= 1.0f) ? 1.0f : 0.0f;
    s.z = (vn.z >= 1.0f) ? 1.0f : 0.0f;
    s.w = (vn.w >= 1.0f) ? 1.0f : 0.0f;
    *(gv4)(og + i) = s;
}

// ---------------------------------------------------------------------------
// Fused GEMM(+GEMM+GEMM) + LIF threshold.
//   z==0: I_h1 = prev_s@W1.T + prev_h2@W3.T + prev_h1@W5.T -> s_h1
//   z==1: I_h2 = prev_h1@W2.T                              -> s_h2
//   z==2: I_m  = prev_h2@W4.T                              -> s_m
// C[i,j] = sum_k A[i,k]*W[j,k]  (x @ W.T: both operands row-major along K).
// 256 threads = 8 waves; 128x128 C tile; wave grid 2(M)x4(N), each wave 64x32
// = 4x2 fragments of v_wmma_f32_16x16x32_bf16.
// ---------------------------------------------------------------------------
__global__ __launch_bounds__(256)
void snn_gemm_lif(const float* __restrict__ prev_s,
                  const float* __restrict__ prev_h1,
                  const float* __restrict__ prev_h2,
                  const float* __restrict__ W1, const float* __restrict__ W2,
                  const float* __restrict__ W3, const float* __restrict__ W4,
                  const float* __restrict__ W5,
                  const float* __restrict__ v_h1,
                  const float* __restrict__ v_h2,
                  const float* __restrict__ v_m,
                  float* __restrict__ out)
{
    __shared__ __bf16 As[BM * LDT];   // 10 KB
    __shared__ __bf16 Ws[BN * LDT];   // 10 KB

    const int z = blockIdx.z;
    gcf A[3]; gcf W[3];
    gcf vmem; gf o; int nT;
    if (z == 0) {
        A[0] = (gcf)(uintptr_t)prev_s;  A[1] = (gcf)(uintptr_t)prev_h2; A[2] = (gcf)(uintptr_t)prev_h1;
        W[0] = (gcf)(uintptr_t)W1;      W[1] = (gcf)(uintptr_t)W3;      W[2] = (gcf)(uintptr_t)W5;
        vmem = (gcf)(uintptr_t)v_h1;
        o = (gf)(uintptr_t)(out + (size_t)1 * B_DIM * N_DIM); nT = 3;
    } else if (z == 1) {
        A[0] = (gcf)(uintptr_t)prev_h1; W[0] = (gcf)(uintptr_t)W2;
        vmem = (gcf)(uintptr_t)v_h2;
        o = (gf)(uintptr_t)(out + (size_t)2 * B_DIM * N_DIM); nT = 1;
    } else {
        A[0] = (gcf)(uintptr_t)prev_h2; W[0] = (gcf)(uintptr_t)W4;
        vmem = (gcf)(uintptr_t)v_m;
        o = (gf)(uintptr_t)(out + (size_t)3 * B_DIM * N_DIM); nT = 1;
    }

    const int colBase = blockIdx.x * BN;
    const int rowBase = blockIdx.y * BM;
    const int t     = threadIdx.x;
    const int wave  = t >> 5;
    const int lane  = t & 31;
    const int half  = lane >> 4;   // 0: lanes 0-15, 1: lanes 16-31
    const int l16   = lane & 15;
    const int waveM = wave >> 2;   // 0..1 -> 64 rows each
    const int waveN = wave & 3;    // 0..3 -> 32 cols each

    v8f acc[4][2];
    #pragma unroll
    for (int i = 0; i < 4; ++i)
        #pragma unroll
        for (int j = 0; j < 2; ++j)
            acc[i][j] = v8f{};

    // cooperative staging indices: thread -> (row 0..127, K half 0/16)
    const int sRow = t >> 1;
    const int sK   = (t & 1) * 16;

    for (int term = 0; term < nT; ++term) {
        gcf Ap = A[term] + (size_t)(rowBase + sRow) * K_DIM + sK;
        gcf Wp = W[term] + (size_t)(colBase + sRow) * K_DIM + sK;

        for (int k0 = 0; k0 < K_DIM; k0 += BK) {
            __syncthreads();  // previous iteration's ds reads complete

            // ---- stage A tile: f32 -> bf16 into LDS ----
            {
                gcf p = Ap + k0;
                Pack8 lo, hi;
                #pragma unroll
                for (int q = 0; q < 2; ++q) {
                    v4f x = *(gcv4)(p + q * 4);
                    lo.e[q*4+0] = (__bf16)x.x; lo.e[q*4+1] = (__bf16)x.y;
                    lo.e[q*4+2] = (__bf16)x.z; lo.e[q*4+3] = (__bf16)x.w;
                }
                #pragma unroll
                for (int q = 0; q < 2; ++q) {
                    v4f x = *(gcv4)(p + 8 + q * 4);
                    hi.e[q*4+0] = (__bf16)x.x; hi.e[q*4+1] = (__bf16)x.y;
                    hi.e[q*4+2] = (__bf16)x.z; hi.e[q*4+3] = (__bf16)x.w;
                }
                __bf16* d = &As[sRow * LDT + sK];
                *(v8bf*)(d)     = lo.v;
                *(v8bf*)(d + 8) = hi.v;
                if (k0 + BK < K_DIM) __builtin_prefetch((const float*)(uintptr_t)(p + BK), 0, 3);
            }
            // ---- stage W tile ----
            {
                gcf p = Wp + k0;
                Pack8 lo, hi;
                #pragma unroll
                for (int q = 0; q < 2; ++q) {
                    v4f x = *(gcv4)(p + q * 4);
                    lo.e[q*4+0] = (__bf16)x.x; lo.e[q*4+1] = (__bf16)x.y;
                    lo.e[q*4+2] = (__bf16)x.z; lo.e[q*4+3] = (__bf16)x.w;
                }
                #pragma unroll
                for (int q = 0; q < 2; ++q) {
                    v4f x = *(gcv4)(p + 8 + q * 4);
                    hi.e[q*4+0] = (__bf16)x.x; hi.e[q*4+1] = (__bf16)x.y;
                    hi.e[q*4+2] = (__bf16)x.z; hi.e[q*4+3] = (__bf16)x.w;
                }
                __bf16* d = &Ws[sRow * LDT + sK];
                *(v8bf*)(d)     = lo.v;
                *(v8bf*)(d + 8) = hi.v;
                if (k0 + BK < K_DIM) __builtin_prefetch((const float*)(uintptr_t)(p + BK), 0, 3);
            }
            __syncthreads();

            // ---- fragment loads (contiguous K order, same for A and B) ----
            // B 32x16: lane holds N=l16; K = 16*half + (0..15), contiguous 32B.
            Frag16 b[2];
            #pragma unroll
            for (int nt = 0; nt < 2; ++nt) {
                const __bf16* bp = &Ws[(waveN * 32 + nt * 16 + l16) * LDT + half * 16];
                b[nt].h[0] = *(const v8bf*)(bp);
                b[nt].h[1] = *(const v8bf*)(bp + 8);
            }
            // A 16x32: lane holds M=l16; K = 16*half + (0..15), contiguous 32B.
            Frag16 a[4];
            #pragma unroll
            for (int mt = 0; mt < 4; ++mt) {
                const __bf16* ap = &As[(waveM * 64 + mt * 16 + l16) * LDT];
                a[mt].h[0] = *(const v8bf*)(ap + half * 16);
                a[mt].h[1] = *(const v8bf*)(ap + half * 16 + 8);
            }

            // ---- 8 WMMAs per K step per wave ----
            #pragma unroll
            for (int mt = 0; mt < 4; ++mt)
                #pragma unroll
                for (int nt = 0; nt < 2; ++nt)
                    acc[mt][nt] = __builtin_amdgcn_wmma_f32_16x16x32_bf16(
                        false, a[mt].f, false, b[nt].f,
                        (short)0, acc[mt][nt], false, false);
        }
    }

    // ---- fused LIF epilogue: spike = (v + 0.05*(I - v)) >= 1 ----
    // C layout: VGPR r -> M = 8*half + r (lanes 0-15: M=0..7; 16-31: M=8..15), N = l16.
    #pragma unroll
    for (int mt = 0; mt < 4; ++mt) {
        #pragma unroll
        for (int nt = 0; nt < 2; ++nt) {
            const int col = colBase + waveN * 32 + nt * 16 + l16;
            #pragma unroll
            for (int r = 0; r < 8; ++r) {
                const int row = rowBase + waveM * 64 + mt * 16 + half * 8 + r;
                const size_t idx = (size_t)row * N_DIM + col;
                const float vv = vmem[idx];
                const float vn = vv + 0.05f * (acc[mt][nt][r] - vv);
                o[idx] = (vn >= 1.0f) ? 1.0f : 0.0f;
            }
        }
    }
}

// ---------------------------------------------------------------------------
extern "C" void kernel_launch(void* const* d_in, const int* in_sizes, int n_in,
                              void* d_out, int out_size, void* d_ws, size_t ws_size,
                              hipStream_t stream)
{
    (void)in_sizes; (void)n_in; (void)out_size; (void)d_ws; (void)ws_size;

    const float* ext     = (const float*)d_in[0];
    const float* prev_s  = (const float*)d_in[1];
    const float* prev_h1 = (const float*)d_in[2];
    const float* prev_h2 = (const float*)d_in[3];
    /* d_in[4] = prev_m: unused by the reference math */
    const float* v_s     = (const float*)d_in[5];
    const float* v_h1    = (const float*)d_in[6];
    const float* v_h2    = (const float*)d_in[7];
    const float* v_m     = (const float*)d_in[8];
    const float* W1      = (const float*)d_in[9];
    const float* W2      = (const float*)d_in[10];
    const float* W3      = (const float*)d_in[11];
    const float* W4      = (const float*)d_in[12];
    const float* W5      = (const float*)d_in[13];
    float* out = (float*)d_out;

    // s_s: 8192*1024 elems / (256 thr * 4 elems) = 8192 blocks
    snn_lif_sensory<<<dim3(8192), dim3(256), 0, stream>>>(ext, v_s, out);

    // fused GEMM+LIF for s_h1, s_h2, s_m
    snn_gemm_lif<<<dim3(N_DIM / BN, B_DIM / BM, 3), dim3(256), 0, stream>>>(
        prev_s, prev_h1, prev_h2, W1, W2, W3, W4, W5, v_h1, v_h2, v_m, out);
}
